// ProtoDINO_44744969289881
// MI455X (gfx1250) — compile-verified, matrix operand order checked by reading, not run on hardware
//
#include <hip/hip_runtime.h>
#include <math.h>

// ---------------------------------------------------------------------------
// ProtoDINO forward for MI455X (gfx1250).  Sizes from the reference:
#define C_CLS   201
#define KPROT   5
#define DIM     768
#define BATCH   64
#define NPATCH  256               // 16x16
#define NROWS   (BATCH * NPATCH)  // 16384  (GEMM M)
#define NCOLS   (C_CLS * KPROT)   // 1005   (GEMM N)
#define NCOLS_P 1024              // padded: 32 column-blocks of 32
#define BN      32                // N tile per block (2 WMMA subtiles)
#define GAMMA_F 0.999f
#define EPS_F   1e-12f
// ---------------------------------------------------------------------------

typedef __attribute__((ext_vector_type(16))) __bf16 v16bf;
typedef __attribute__((ext_vector_type(8)))  float  v8f;
typedef __attribute__((ext_vector_type(4)))  unsigned int v4u;
typedef __attribute__((ext_vector_type(8)))  int    v8i_;
typedef __attribute__((ext_vector_type(4)))  int    v4i_;

static __device__ __forceinline__ __bf16 f2bf(float f) {
    unsigned int u = __builtin_bit_cast(unsigned int, f);
    unsigned int lsb = (u >> 16) & 1u;
    u += 0x7fffu + lsb;                       // round-to-nearest-even
    unsigned short h = (unsigned short)(u >> 16);
    return __builtin_bit_cast(__bf16, h);
}
static __device__ __forceinline__ float bf2f(__bf16 b) {
    unsigned short h = __builtin_bit_cast(unsigned short, b);
    unsigned int u = ((unsigned int)h) << 16;
    return __builtin_bit_cast(float, u);
}
static __device__ __forceinline__ int row_class(int n, const int* labels,
                                                const unsigned char* masks) {
    int b = n >> 8, pi = n & 255;
    return masks[(size_t)b * NPATCH + pi] ? labels[b] : (C_CLS - 1);
}

// --------------------------- 1. row L2-normalize -> bf16 -------------------
__global__ __launch_bounds__(256)
void k_rownorm_bf16(const float* __restrict__ src, __bf16* __restrict__ dst,
                    int valid_rows) {
    int row = blockIdx.x;
    __shared__ float red[256];
    size_t base = (size_t)row * DIM;
    int t = threadIdx.x;
    float x0 = 0.f, x1 = 0.f, x2 = 0.f;
    bool valid = row < valid_rows;
    if (valid) {
        x0 = src[base + t];
        x1 = src[base + t + 256];
        x2 = src[base + t + 512];
    }
    red[t] = x0 * x0 + x1 * x1 + x2 * x2;
    __syncthreads();
    for (int off = 128; off > 0; off >>= 1) {
        if (t < off) red[t] += red[t + off];
        __syncthreads();
    }
    float inv = valid ? 1.0f / (sqrtf(red[0]) + EPS_F) : 0.f;
    dst[base + t]       = f2bf(x0 * inv);
    dst[base + t + 256] = f2bf(x1 * inv);
    dst[base + t + 512] = f2bf(x2 * inv);
}

// --------------------------- 2. WMMA GEMM ----------------------------------
// logits[m, n] = sum_d tokn[m,d] * pron[n,d]
// block = 256 threads = 8 waves; block tile = 128 M x 32 N.
// B tile (32 x 768 bf16, 48 KB) staged in LDS via the Tensor Data Mover;
// each wave issues 2 WMMAs per A-fragment (two 16-wide N subtiles).
__global__ __launch_bounds__(256)
void k_gemm_wmma(const __bf16* __restrict__ tokn, const __bf16* __restrict__ pron,
                 float* __restrict__ out) {
    __shared__ __bf16 Bsh[BN * DIM];          // 48 KB

    int wave = threadIdx.x >> 5;
    int lane = threadIdx.x & 31;
    int lr = lane & 15;          // row within A-tile / col within B-tile
    int lh = lane >> 4;          // which 16-wide K half this lane holds
    int m0 = blockIdx.x * 128 + wave * 16;
    int n0 = blockIdx.y * BN;

    // ---- stage B tile: global[n0 .. n0+31][0..767] -> LDS -----------------
#if __has_builtin(__builtin_amdgcn_tensor_load_to_lds) && __has_builtin(__builtin_amdgcn_s_wait_tensorcnt)
    if (wave == 0) {
        unsigned long long ga =
            (unsigned long long)(const void*)(pron + (size_t)n0 * DIM);
        unsigned lds = (unsigned)(unsigned long long)(const void*)&Bsh[0];
        // D# group0: count=1, lds_addr, global_addr, type=2 ("image")
        v4u g0;
        g0.x = 1u;                                        // count = 1 valid D#
        g0.y = lds;                                       // lds_addr [63:32]
        g0.z = (unsigned)(ga & 0xffffffffu);              // global_addr lo
        g0.w = ((unsigned)(ga >> 32) & 0x01ffffffu) | (2u << 30);  // hi | type=2
        // D# group1: data_size=2B, tensor 768 x 1024 (stride 768), tile 768 x 32
        v8i_ g1;
        g1[0] = (int)(1u << 16);          // data_size=1 (2 bytes); mask/flags=0
        g1[1] = (int)(768u << 16);        // tensor_dim0 = 768  (bits 79:48)
        g1[2] = (int)(1024u << 16);       // tensor_dim1 = 1024 (bits 111:80)
        g1[3] = (int)(768u << 16);        // tile_dim0   = 768  (bits 127:112)
        g1[4] = (int)BN;                  // tile_dim1   = 32   (bits 143:128)
        g1[5] = 768;                      // tensor_dim0_stride (bits 207:160)
        g1[6] = 0;
        g1[7] = 0;
        v4i_ gz = {0, 0, 0, 0};
#if __clang_major__ >= 23
        v8i_ gz8 = {0, 0, 0, 0, 0, 0, 0, 0};
        __builtin_amdgcn_tensor_load_to_lds(g0, g1, gz, gz, gz8, 0);
#else
        __builtin_amdgcn_tensor_load_to_lds(g0, g1, gz, gz, 0);
#endif
        __builtin_amdgcn_s_wait_tensorcnt(0);
    }
#else
    // fallback: cooperative copy, 16B per transfer
    {
        const uint4* src = (const uint4*)(pron + (size_t)n0 * DIM);
        uint4* dst = (uint4*)&Bsh[0];
        const int nvec = BN * DIM * 2 / 16;   // 3072
        for (int i = threadIdx.x; i < nvec; i += 256) dst[i] = src[i];
    }
#endif
    __syncthreads();

    // ---- K loop: 24 steps of 32; 1 A load + 2 LDS B frags + 2 WMMA --------
    const __bf16* aptr = tokn + (size_t)(m0 + lr) * DIM + lh * 16;
    const __bf16* b0p  = &Bsh[(size_t)lr * DIM + lh * 16];
    const __bf16* b1p  = &Bsh[(size_t)(16 + lr) * DIM + lh * 16];

    v8f acc0 = {}, acc1 = {};
#pragma unroll 4
    for (int k0 = 0; k0 < DIM; k0 += 32) {
        if (k0 + 64 < DIM)                        // global_prefetch_b8 path
            __builtin_prefetch(aptr + k0 + 64, 0, 3);
        v16bf a  = *(const v16bf*)(aptr + k0);
        v16bf b0 = *(const v16bf*)(b0p + k0);
        v16bf b1 = *(const v16bf*)(b1p + k0);
        acc0 = __builtin_amdgcn_wmma_f32_16x16x32_bf16(
                   false, a, false, b0, (short)0, acc0, false, false);
        acc1 = __builtin_amdgcn_wmma_f32_16x16x32_bf16(
                   false, a, false, b1, (short)0, acc1, false, false);
    }

    // C/D layout: VGPR r, lanes 0-15 -> M=r, lanes 16-31 -> M=r+8; col = lane&15
    int rowbase = m0 + lh * 8;
    int col0 = n0 + lr;
    int col1 = n0 + 16 + lr;
    if (col0 < NCOLS) {
#pragma unroll
        for (int r = 0; r < 8; ++r)
            out[(size_t)(rowbase + r) * NCOLS + col0] = acc0[r];
    }
    if (col1 < NCOLS) {
#pragma unroll
        for (int r = 0; r < 8; ++r)
            out[(size_t)(rowbase + r) * NCOLS + col1] = acc1[r];
    }
}

// --------------------------- 3. mean over patches --------------------------
__global__ __launch_bounds__(256)
void k_meanpool(const float* __restrict__ logits, float* __restrict__ img) {
    int b = blockIdx.x;
    int ck = blockIdx.y * 256 + threadIdx.x;
    if (ck >= NCOLS) return;
    const float* p = logits + (size_t)b * NPATCH * NCOLS + ck;
    float s = 0.f;
    for (int n = 0; n < NPATCH; ++n) s += p[(size_t)n * NCOLS];
    img[(size_t)b * NCOLS + ck] = s * (1.0f / NPATCH);
}

// --------------------------- 4. sa classification head ---------------------
__global__ __launch_bounds__(256)
void k_class_logits(const float* __restrict__ img, const float* __restrict__ sa,
                    const float* __restrict__ scale_p, float* __restrict__ out) {
    int b = blockIdx.x;
    int c = threadIdx.x;
    if (c >= C_CLS - 1) return;
    float s[KPROT];
    float mx = sa[c * KPROT];
    for (int k = 1; k < KPROT; ++k) mx = fmaxf(mx, sa[c * KPROT + k]);
    float sum = 0.f;
    for (int k = 0; k < KPROT; ++k) { s[k] = __expf(sa[c * KPROT + k] - mx); sum += s[k]; }
    float invs = (float)KPROT / sum;           // softmax * K
    float acc = 0.f;
    for (int k = 0; k < KPROT; ++k)
        acc += img[(size_t)b * NCOLS + c * KPROT + k] * s[k] * invs;
    out[(size_t)b * (C_CLS - 1) + c] = scale_p[0] * acc;
}

// --------------------------- 5. Sinkhorn column-sum pass -------------------
// Q_all[c,n,:] is nonzero only for pl[n]==c, so each iteration only needs the
// 5 per-class column sums; rows are reconstructed on the fly.
static __device__ __forceinline__ void sk_row(const float* __restrict__ logits,
                                              const float* __restrict__ CS,
                                              int n, int c, int iters, float* q) {
    const float* L = logits + (size_t)n * NCOLS + c * KPROT;
    float mx = L[0];
    for (int k = 1; k < KPROT; ++k) mx = fmaxf(mx, L[k]);
    float s = 0.f;
    for (int k = 0; k < KPROT; ++k) { q[k] = __expf(L[k] - mx); s += q[k]; }
    float invs = 1.0f / s;
    for (int k = 0; k < KPROT; ++k) q[k] *= invs;
    for (int tt = 0; tt < iters; ++tt) {
        float rs = 0.f;
        const float* cs = CS + (size_t)tt * NCOLS + c * KPROT;
        for (int k = 0; k < KPROT; ++k) { q[k] /= (cs[k] + EPS_F); rs += q[k]; }
        float ir = 1.0f / (rs + EPS_F);
        for (int k = 0; k < KPROT; ++k) q[k] *= ir;
    }
}

__global__ __launch_bounds__(256)
void k_sk_pass(const float* __restrict__ logits, const int* __restrict__ labels,
               const unsigned char* __restrict__ masks, float* __restrict__ CS,
               int* __restrict__ has_class, int t) {
    int c = blockIdx.x;
    int tid = threadIdx.x;
    float acc[KPROT] = {0.f, 0.f, 0.f, 0.f, 0.f};
    int found = 0;
    for (int n = tid; n < NROWS; n += 256) {
        if (row_class(n, labels, masks) != c) continue;
        found = 1;
        float q[KPROT];
        sk_row(logits, CS, n, c, t, q);
        for (int k = 0; k < KPROT; ++k) acc[k] += q[k];
    }
    __shared__ float red[256];
    __shared__ int   fr[256];
    fr[tid] = found;
    __syncthreads();
    for (int off = 128; off > 0; off >>= 1) {
        if (tid < off) fr[tid] |= fr[tid + off];
        __syncthreads();
    }
    if (t == 0 && tid == 0) has_class[c] = fr[0];
    for (int k = 0; k < KPROT; ++k) {
        __syncthreads();
        red[tid] = acc[k];
        __syncthreads();
        for (int off = 128; off > 0; off >>= 1) {
            if (tid < off) red[tid] += red[tid + off];
            __syncthreads();
        }
        if (tid == 0) CS[(size_t)t * NCOLS + c * KPROT + k] = red[0];
    }
}

// --------------------------- 6. finalize rows ------------------------------
__global__ __launch_bounds__(256)
void k_finalize(const float* __restrict__ logits, const int* __restrict__ labels,
                const unsigned char* __restrict__ masks, const float* __restrict__ CS,
                float* __restrict__ w, float* __restrict__ out_part,
                float* __restrict__ out_pl) {
    int n = blockIdx.x * 256 + threadIdx.x;
    if (n >= NROWS) return;
    int c = row_class(n, labels, masks);
    out_pl[n] = (float)c;
    float q[KPROT];
    sk_row(logits, CS, n, c, 3, q);
    int best = 0;
    float bv = q[0];
    for (int k = 1; k < KPROT; ++k) if (q[k] > bv) { bv = q[k]; best = k; }
    out_part[n] = (float)best;
    for (int k = 0; k < KPROT; ++k) w[(size_t)n * KPROT + k] = q[k];
}

// --------------------------- 7. prototype EMA update -----------------------
__global__ __launch_bounds__(256)
void k_proto_update(const __bf16* __restrict__ tokn, const float* __restrict__ w,
                    const int* __restrict__ labels, const unsigned char* __restrict__ masks,
                    const int* __restrict__ has_class, const float* __restrict__ proto,
                    float* __restrict__ out_np) {
    int c = blockIdx.x;
    int t = threadIdx.x;
    float acc[KPROT][3];
#pragma unroll
    for (int k = 0; k < KPROT; ++k)
        for (int j = 0; j < 3; ++j) acc[k][j] = 0.f;

    for (int n = 0; n < NROWS; ++n) {
        if (row_class(n, labels, masks) != c) continue;
        float wk[KPROT];
        for (int k = 0; k < KPROT; ++k) wk[k] = w[(size_t)n * KPROT + k];
        const __bf16* tp = tokn + (size_t)n * DIM + t;
#pragma unroll
        for (int j = 0; j < 3; ++j) {
            float x = bf2f(tp[j * 256]);
            for (int k = 0; k < KPROT; ++k) acc[k][j] += wk[k] * x;
        }
    }
    int hc = has_class[c];
#pragma unroll
    for (int k = 0; k < KPROT; ++k)
        for (int j = 0; j < 3; ++j) {
            size_t idx = ((size_t)c * KPROT + k) * DIM + t + j * 256;
            float op = proto[idx];
            out_np[idx] = hc ? (GAMMA_F * op + (1.0f - GAMMA_F) * acc[k][j]) : op;
        }
}

// ===========================================================================
extern "C" void kernel_launch(void* const* d_in, const int* in_sizes, int n_in,
                              void* d_out, int out_size, void* d_ws, size_t ws_size,
                              hipStream_t stream) {
    (void)in_sizes; (void)n_in; (void)out_size; (void)ws_size;

    const float*         tok    = (const float*)d_in[0];          // [64,256,768]
    const float*         proto  = (const float*)d_in[1];          // [201,5,768]
    const float*         sa     = (const float*)d_in[2];          // [200,5]
    const float*         scale  = (const float*)d_in[3];          // scalar
    const int*           labels = (const int*)d_in[4];            // [64]
    const unsigned char* masks  = (const unsigned char*)d_in[5];  // [64,16,16] bool

    // outputs (flat float, in reference return order)
    float* o_logits = (float*)d_out;                              // [B,N,C,K]
    float* o_img    = o_logits + (size_t)NROWS * NCOLS;           // [B,C,K]
    float* o_cls    = o_img + (size_t)BATCH * NCOLS;              // [B,C-1]
    float* o_part   = o_cls + (size_t)BATCH * (C_CLS - 1);        // [B,N]
    float* o_newp   = o_part + (size_t)NROWS;                     // [C,K,D]
    float* o_pl     = o_newp + (size_t)C_CLS * KPROT * DIM;       // [B,H,W]

    // workspace layout (256B aligned)
    char* wsb = (char*)d_ws;
    size_t off = 0;
    auto take = [&](size_t bytes) {
        void* p = wsb + off;
        off = (off + bytes + 255) & ~(size_t)255;
        return p;
    };
    __bf16* tokn = (__bf16*)take((size_t)NROWS * DIM * sizeof(__bf16));
    __bf16* pron = (__bf16*)take((size_t)NCOLS_P * DIM * sizeof(__bf16));
    float*  CS   = (float*)take((size_t)3 * NCOLS * sizeof(float));
    int*    hasc = (int*)take((size_t)C_CLS * sizeof(int));
    float*  wbuf = (float*)take((size_t)NROWS * KPROT * sizeof(float));

    // 1. normalize tokens & prototypes to bf16 (pad proto rows to 1024)
    k_rownorm_bf16<<<NROWS, 256, 0, stream>>>(tok, tokn, NROWS);
    k_rownorm_bf16<<<NCOLS_P, 256, 0, stream>>>(proto, pron, NCOLS);

    // 2. cosine logits GEMM: TDM-staged B tile + v_wmma_f32_16x16x32_bf16
    dim3 ggrid(NROWS / 128, NCOLS_P / BN);   // (128, 32)
    k_gemm_wmma<<<ggrid, 256, 0, stream>>>(tokn, pron, o_logits);

    // 3. mean pooling  4. class head
    k_meanpool<<<dim3(BATCH, (NCOLS + 255) / 256), 256, 0, stream>>>(o_logits, o_img);
    k_class_logits<<<BATCH, 256, 0, stream>>>(o_img, sa, scale, o_cls);

    // 5. three Sinkhorn column-sum passes (CS[t] for t = 0,1,2)
    for (int t = 0; t < 3; ++t)
        k_sk_pass<<<C_CLS, 256, 0, stream>>>(o_logits, labels, masks, CS, hasc, t);

    // 6. per-row finalize: argmax, weights, pseudo labels
    k_finalize<<<NROWS / 256, 256, 0, stream>>>(o_logits, labels, masks, CS,
                                                wbuf, o_part, o_pl);

    // 7. prototype EMA update
    k_proto_update<<<C_CLS, 256, 0, stream>>>(tokn, wbuf, labels, masks, hasc,
                                              proto, o_newp);
}
